// ContactPointTrans_962072675001
// MI455X (gfx1250) — compile-verified, matrix operand order checked by reading, not run on hardware
//
#include <hip/hip_runtime.h>

// ---------------- WMMA fragment types (wave32, CDNA5) ----------------
typedef __attribute__((ext_vector_type(16))) _Float16 v16h;
typedef __attribute__((ext_vector_type(8)))  float    v8f;
typedef __attribute__((ext_vector_type(4)))  unsigned int u32x4;
typedef __attribute__((ext_vector_type(8)))  int i32x8;
typedef __attribute__((ext_vector_type(4)))  int i32x4;

#if __has_builtin(__builtin_amdgcn_tensor_load_to_lds)
#define USE_TDM 1
#else
#define USE_TDM 0
#endif

#define BM 64
#define BN 64
#define BK 32

// =====================================================================
// GEMM: C[M,N] = A[M,K] * W[K,N] (+bias) (+relu), f16 WMMA, f32 accum.
// 256 threads = 8 waves. Block tile 64x64; each wave owns two 16x16
// output tiles (rows +0 / +32, same cols) sharing one B fragment.
// A tile (64x32 f32) is DMA'd into LDS by the Tensor Data Mover;
// B tile is staged transposed in f16 so fragment reads are contiguous
// (2 x ds_load_b128 per fragment).
// =====================================================================
__global__ __launch_bounds__(256) void gemm_wmma_kernel(
    const float* __restrict__ A, const float* __restrict__ W,
    const float* __restrict__ bias, float* __restrict__ C,
    int M, int K, int N, int relu)
{
  __shared__ float    sA32[BM][BK];   // 8 KB, row-major (TDM layout)
  __shared__ _Float16 sBt[BN][BK];    // 4 KB, transposed W tile
  const int tid  = threadIdx.x;
  const int lane = tid & 31;
  const int wave = tid >> 5;
  const int rb0  = blockIdx.y * BM;
  const int cb0  = blockIdx.x * BN;
  const int tr   = (wave & 1) << 4;
  const int tc   = (wave >> 1) << 4;
  const int lid  = lane & 15;
  const int koff = (lane >> 4) << 3;   // 0 / 8 : K sub-group per ISA layout
  v8f acc0 = {}; v8f acc1 = {};

  for (int k0 = 0; k0 < K; k0 += BK) {
    // ---------------- A tile via Tensor Data Mover ----------------
#if USE_TDM
    if (wave == 0) {
      unsigned long long ga = (unsigned long long)(uintptr_t)(A + (size_t)rb0 * K + k0);
      unsigned ldsa = (unsigned)(uintptr_t)(void*)&sA32[0][0];  // LDS offset = addr[31:0]
      unsigned td0 = (unsigned)(K - k0);     // remaining elems in row (HW zero-fills OOB)
      unsigned td1 = (unsigned)(M - rb0);    // remaining rows
      unsigned long long st0 = (unsigned long long)K;
      u32x4 g0;
      g0[0] = 1u;                                          // count=1 valid descriptor
      g0[1] = ldsa;                                        // lds_addr
      g0[2] = (unsigned)(ga & 0xFFFFFFFFu);                // global_addr[31:0]
      g0[3] = (unsigned)((ga >> 32) & 0x1FFFFFFu) | (2u << 30);  // addr[56:32] | type=2
      i32x8 g1;
      g1[0] = (int)(2u << 16);                             // data_size = 4 bytes
      g1[1] = (int)((td0 & 0xFFFFu) << 16);                // tensor_dim0[15:0]
      g1[2] = (int)((td0 >> 16) | ((td1 & 0xFFFFu) << 16));// dim0[31:16] | dim1[15:0]
      g1[3] = (int)((td1 >> 16) | ((unsigned)BK << 16));   // dim1[31:16] | tile_dim0=32
      g1[4] = (int)BM;                                     // tile_dim1=64, tile_dim2=0
      g1[5] = (int)(st0 & 0xFFFFFFFFu);                    // dim0 stride[31:0]
      g1[6] = (int)((st0 >> 32) & 0xFFFFu);                // stride[47:32] | dim1_stride=0
      g1[7] = 0;
      i32x4 z4 = {0, 0, 0, 0};
      i32x8 z8 = {0, 0, 0, 0, 0, 0, 0, 0};
      __builtin_amdgcn_tensor_load_to_lds(g0, g1, z4, z4, z8, 0);
      __builtin_amdgcn_s_wait_tensorcnt(0);
    }
#else
    for (int i = tid; i < BM * BK; i += 256) {
      int r = i >> 5, kk = i & 31;
      int gr = rb0 + r, gk = k0 + kk;
      sA32[r][kk] = (gr < M && gk < K) ? A[(size_t)gr * K + gk] : 0.f;
    }
#endif
    // ---------------- B tile (transposed, f16) ----------------
    if (cb0 + BN <= N && k0 + BK <= K) {       // fast interior path
      for (int i = tid; i < BK * BN; i += 256) {
        int kk = i >> 6, cc = i & 63;
        sBt[cc][kk] = (_Float16)W[(size_t)(k0 + kk) * N + (cb0 + cc)];
      }
    } else {
      for (int i = tid; i < BK * BN; i += 256) {
        int kk = i >> 6, cc = i & 63;
        int gk = k0 + kk, gc = cb0 + cc;
        float v = (gk < K && gc < N) ? W[(size_t)gk * N + gc] : 0.f;
        sBt[cc][kk] = (_Float16)v;
      }
    }
    // speculative prefetch of next W tile (global_prefetch_b8)
    if (k0 + BK < K) {
      int pc = cb0 + (tid & 63), pk = k0 + BK + (tid >> 6);
      if (pc < N && pk < K) __builtin_prefetch(&W[(size_t)pk * N + pc], 0, 0);
    }
    __syncthreads();

    // ---------------- fragments (contiguous LDS reads) ----------------
    const float*    ar0 = &sA32[tr + lid][0];
    const float*    ar1 = &sA32[tr + 32 + lid][0];
    const _Float16* br  = &sBt[tc + lid][0];
    v16h a0, a1, bf;
#pragma unroll
    for (int h = 0; h < 8; ++h) {
      a0[h]     = (_Float16)ar0[koff + h];
      a0[h + 8] = (_Float16)ar0[16 + koff + h];
      a1[h]     = (_Float16)ar1[koff + h];
      a1[h + 8] = (_Float16)ar1[16 + koff + h];
      bf[h]     = br[koff + h];
      bf[h + 8] = br[16 + koff + h];
    }
    acc0 = __builtin_amdgcn_wmma_f32_16x16x32_f16(false, a0, false, bf,
                                                  (short)0, acc0, false, false);
    acc1 = __builtin_amdgcn_wmma_f32_16x16x32_f16(false, a1, false, bf,
                                                  (short)0, acc1, false, false);
    __syncthreads();
  }

  // D layout: VGPR e -> M = e (+8 for lanes>=16), N = lane&15
  const int ocol  = cb0 + tc + lid;
  const int orow0 = rb0 + tr + koff;
  const float bv  = (bias != nullptr && ocol < N) ? bias[ocol] : 0.f;
#pragma unroll
  for (int e = 0; e < 8; ++e) {
    int r0 = orow0 + e;
    if (r0 < M && ocol < N) {
      float o = acc0[e] + bv;
      if (relu) o = fmaxf(o, 0.f);
      C[(size_t)r0 * N + ocol] = o;
    }
    int r1 = r0 + 32;
    if (r1 < M && ocol < N) {
      float o = acc1[e] + bv;
      if (relu) o = fmaxf(o, 0.f);
      C[(size_t)r1 * N + ocol] = o;
    }
  }
}

// ============================ BatchNorm ==============================
__global__ __launch_bounds__(256) void bn_stats_kernel(
    const float* __restrict__ x, int M, int C, float* __restrict__ mv)
{
  int c = blockIdx.x;
  __shared__ float s1[256], s2[256];
  float a = 0.f, b = 0.f;
  for (int r = threadIdx.x; r < M; r += 256) {
    float v = x[(size_t)r * C + c];
    a += v; b += v * v;
  }
  s1[threadIdx.x] = a; s2[threadIdx.x] = b;
  __syncthreads();
  for (int st = 128; st > 0; st >>= 1) {
    if (threadIdx.x < st) {
      s1[threadIdx.x] += s1[threadIdx.x + st];
      s2[threadIdx.x] += s2[threadIdx.x + st];
    }
    __syncthreads();
  }
  if (threadIdx.x == 0) {
    float m = s1[0] / (float)M;
    mv[c] = m;
    mv[C + c] = fmaxf(s2[0] / (float)M - m * m, 0.f);
  }
}

__global__ __launch_bounds__(256) void bn_apply_kernel(
    float* __restrict__ x, const float* __restrict__ mv,
    const float* __restrict__ g, const float* __restrict__ b,
    int M, int C, int relu)
{
  long long i = (long long)blockIdx.x * 256 + threadIdx.x;
  if (i >= (long long)M * C) return;
  int c = (int)(i % C);
  float sc = g[c] * rsqrtf(mv[C + c] + 1e-5f);
  float v = (x[i] - mv[c]) * sc + b[c];
  x[i] = relu ? fmaxf(v, 0.f) : v;
}

// ============================ KNN (top-k scan) =======================
template <int KN>
__global__ __launch_bounds__(256) void knn_kernel(
    const float* __restrict__ q, const float* __restrict__ r,
    int* __restrict__ idx, int B, int m, int n)
{
  long long i = (long long)blockIdx.x * 256 + threadIdx.x;
  if (i >= (long long)B * m) return;
  int b = (int)(i / m), qi = (int)(i % m);
  const float* qp = q + ((long long)b * m + qi) * 3;
  float qx = qp[0], qy = qp[1], qz = qp[2];
  const float* rp = r + (long long)b * n * 3;
  float bd[KN]; int bi[KN];
#pragma unroll
  for (int t = 0; t < KN; ++t) { bd[t] = 3.4e38f; bi[t] = 0; }
  for (int j = 0; j < n; ++j) {
    float dx = rp[j * 3] - qx, dy = rp[j * 3 + 1] - qy, dz = rp[j * 3 + 2] - qz;
    float d2 = dx * dx + dy * dy + dz * dz;
    if (d2 < bd[KN - 1]) {
      bd[KN - 1] = d2; bi[KN - 1] = j;
#pragma unroll
      for (int t = KN - 1; t > 0; --t) {
        if (bd[t] < bd[t - 1]) {
          float tf = bd[t]; bd[t] = bd[t - 1]; bd[t - 1] = tf;
          int ti = bi[t]; bi[t] = bi[t - 1]; bi[t - 1] = ti;
        }
      }
    }
  }
  int* op = idx + ((long long)b * m + qi) * KN;
#pragma unroll
  for (int t = 0; t < KN; ++t) op[t] = bi[t];
}

// ===================== Furthest Point Sampling =======================
__global__ __launch_bounds__(256) void fps_kernel(
    const float* __restrict__ p, int* __restrict__ sidx, int B, int n, int m)
{
  int b = blockIdx.x;
  const float* pb = p + (size_t)b * n * 3;
  __shared__ float sd[4096];
  __shared__ float rv[256];
  __shared__ int   ra[256];
  __shared__ int   slast;
  for (int i = threadIdx.x; i < n; i += 256) sd[i] = 1e10f;
  if (threadIdx.x == 0) { sidx[(size_t)b * m] = 0; slast = 0; }
  __syncthreads();
  for (int it = 1; it < m; ++it) {
    int last = slast;
    float lx = pb[last * 3], ly = pb[last * 3 + 1], lz = pb[last * 3 + 2];
    float best = -1.f; int barg = 0;
    for (int i = threadIdx.x; i < n; i += 256) {
      float dx = pb[i * 3] - lx, dy = pb[i * 3 + 1] - ly, dz = pb[i * 3 + 2] - lz;
      float d = dx * dx + dy * dy + dz * dz;
      float nd = fminf(sd[i], d);
      sd[i] = nd;
      if (nd > best) { best = nd; barg = i; }
    }
    rv[threadIdx.x] = best; ra[threadIdx.x] = barg;
    __syncthreads();
    for (int st = 128; st > 0; st >>= 1) {
      if (threadIdx.x < st && rv[threadIdx.x + st] > rv[threadIdx.x]) {
        rv[threadIdx.x] = rv[threadIdx.x + st];
        ra[threadIdx.x] = ra[threadIdx.x + st];
      }
      __syncthreads();
    }
    if (threadIdx.x == 0) { slast = ra[0]; sidx[(size_t)b * m + it] = ra[0]; }
    __syncthreads();
  }
}

// ========================= gather / grouping =========================
__global__ __launch_bounds__(256) void gather_kernel(
    const float* __restrict__ src, const int* __restrict__ idx,
    float* __restrict__ out, int B, int m, int k, int n, int C)
{
  long long i = (long long)blockIdx.x * 256 + threadIdx.x;
  long long tot = (long long)B * m * k * C;
  if (i >= tot) return;
  int c = (int)(i % C);
  long long t = i / C;
  int s = (int)(t % k); t /= k;
  int qm = (int)(t % m);
  int b = (int)(t / m);
  int id = idx[((long long)b * m + qm) * k + s];
  out[i] = src[((long long)b * n + id) * C + c];
}

// out[b,qm,s, 0:3] = p[idx]-npos ; out[b,qm,s, 3:3+c] = x[idx]
__global__ __launch_bounds__(256) void group_cat_kernel(
    const float* __restrict__ p, const float* __restrict__ x,
    const float* __restrict__ npos, const int* __restrict__ nidx,
    float* __restrict__ out, int B, int m, int k, int n, int c)
{
  int C = 3 + c;
  long long i = (long long)blockIdx.x * 256 + threadIdx.x;
  long long tot = (long long)B * m * k * C;
  if (i >= tot) return;
  int cc = (int)(i % C);
  long long t = i / C;
  int s = (int)(t % k); t /= k;
  int qm = (int)(t % m);
  int b = (int)(t / m);
  int id = nidx[((long long)b * m + qm) * k + s];
  if (cc < 3)
    out[i] = p[((long long)b * n + id) * 3 + cc] - npos[((long long)b * m + qm) * 3 + cc];
  else
    out[i] = x[((long long)b * n + id) * c + (cc - 3)];
}

__global__ __launch_bounds__(256) void maxpool_ns_kernel(
    const float* __restrict__ x, float* __restrict__ out, int rows, int ns, int c)
{
  long long i = (long long)blockIdx.x * 256 + threadIdx.x;
  if (i >= (long long)rows * c) return;
  int ch = (int)(i % c);
  long long r = i / c;
  float mx = -3.4e38f;
  for (int s = 0; s < ns; ++s) mx = fmaxf(mx, x[(r * ns + s) * c + ch]);
  out[i] = mx;
}

// kg <- kg - q (broadcast over ns) + pr
__global__ __launch_bounds__(256) void qkp_combine_kernel(
    float* __restrict__ kg, const float* __restrict__ q,
    const float* __restrict__ pr, int rows, int ns, int c)
{
  long long i = (long long)blockIdx.x * 256 + threadIdx.x;
  if (i >= (long long)rows * ns * c) return;
  int ch = (int)(i % c);
  long long r = i / ((long long)c * ns);
  kg[i] = kg[i] - q[r * c + ch] + pr[i];
}

__global__ __launch_bounds__(256) void softmax_ns_kernel(
    float* __restrict__ w, int rows, int ns, int cs)
{
  long long i = (long long)blockIdx.x * 256 + threadIdx.x;
  if (i >= (long long)rows * cs) return;
  int ch = (int)(i % cs);
  long long r = i / cs;
  float* base = w + (r * ns) * cs + ch;
  float mx = -3.4e38f;
  for (int s = 0; s < ns; ++s) mx = fmaxf(mx, base[(size_t)s * cs]);
  float sum = 0.f;
  for (int s = 0; s < ns; ++s) {
    float e = __expf(base[(size_t)s * cs] - mx);
    base[(size_t)s * cs] = e;
    sum += e;
  }
  float inv = 1.f / sum;
  for (int s = 0; s < ns; ++s) base[(size_t)s * cs] *= inv;
}

// out[r,ch] = sum_s (vg+pr)[r,s,ch] * w[r,s,ch % cs]
__global__ __launch_bounds__(256) void attn_out_kernel(
    const float* __restrict__ vg, const float* __restrict__ pr,
    const float* __restrict__ w, float* __restrict__ out,
    int rows, int ns, int c, int cs)
{
  long long i = (long long)blockIdx.x * 256 + threadIdx.x;
  if (i >= (long long)rows * c) return;
  int ch = (int)(i % c);
  long long r = i / c;
  int wc = ch % cs;
  float acc = 0.f;
  for (int s = 0; s < ns; ++s) {
    long long e = (r * ns + s) * (long long)c + ch;
    acc += (vg[e] + pr[e]) * w[(r * ns + s) * (long long)cs + wc];
  }
  out[i] = acc;
}

__global__ __launch_bounds__(256) void add_kernel(
    float* __restrict__ y, const float* __restrict__ a, long long n, int relu)
{
  long long i = (long long)blockIdx.x * 256 + threadIdx.x;
  if (i >= n) return;
  float v = y[i] + a[i];
  y[i] = relu ? fmaxf(v, 0.f) : v;
}

__global__ __launch_bounds__(256) void mean_axis1_kernel(
    const float* __restrict__ x, float* __restrict__ out, int B, int n, int C)
{
  int i = blockIdx.x * 256 + threadIdx.x;
  if (i >= B * C) return;
  int b = i / C, c = i % C;
  float s = 0.f;
  for (int r = 0; r < n; ++r) s += x[((long long)b * n + r) * C + c];
  out[i] = s / (float)n;
}

// 3-NN inverse-distance interpolation, fused knn3 + weighted gather
__global__ __launch_bounds__(256) void interp_kernel(
    const float* __restrict__ pf, const float* __restrict__ pc,
    const float* __restrict__ xc, float* __restrict__ out,
    int B, int nf, int nc, int C)
{
  long long i = (long long)blockIdx.x * 256 + threadIdx.x;
  if (i >= (long long)B * nf) return;
  int b = (int)(i / nf), qi = (int)(i % nf);
  const float* qp = pf + ((long long)b * nf + qi) * 3;
  float qx = qp[0], qy = qp[1], qz = qp[2];
  const float* rp = pc + (long long)b * nc * 3;
  float bd[3] = {3.4e38f, 3.4e38f, 3.4e38f};
  int bi3[3] = {0, 0, 0};
  for (int j = 0; j < nc; ++j) {
    float dx = rp[j * 3] - qx, dy = rp[j * 3 + 1] - qy, dz = rp[j * 3 + 2] - qz;
    float d2 = dx * dx + dy * dy + dz * dz;
    if (d2 < bd[2]) {
      bd[2] = d2; bi3[2] = j;
      if (bd[2] < bd[1]) { float tf = bd[2]; bd[2] = bd[1]; bd[1] = tf; int ti = bi3[2]; bi3[2] = bi3[1]; bi3[1] = ti; }
      if (bd[1] < bd[0]) { float tf = bd[1]; bd[1] = bd[0]; bd[0] = tf; int ti = bi3[1]; bi3[1] = bi3[0]; bi3[0] = ti; }
    }
  }
  float w[3]; float ws = 0.f;
  for (int t = 0; t < 3; ++t) { w[t] = 1.f / (sqrtf(fmaxf(bd[t], 0.f)) + 1e-8f); ws += w[t]; }
  for (int t = 0; t < 3; ++t) w[t] /= ws;
  const float* x0 = xc + ((long long)b * nc + bi3[0]) * C;
  const float* x1 = xc + ((long long)b * nc + bi3[1]) * C;
  const float* x2 = xc + ((long long)b * nc + bi3[2]) * C;
  float* o = out + ((long long)b * nf + qi) * C;
  for (int cc = 0; cc < C; ++cc) o[cc] = w[0] * x0[cc] + w[1] * x1[cc] + w[2] * x2[cc];
}

// x0 = concat(p(3), x(1), pf(64)) -> (B,N,68)
__global__ __launch_bounds__(256) void build_x0_kernel(
    const float* __restrict__ p, const float* __restrict__ xin,
    const float* __restrict__ pf, float* __restrict__ out, int B, int n)
{
  const int C = 68;
  long long i = (long long)blockIdx.x * 256 + threadIdx.x;
  if (i >= (long long)B * n * C) return;
  int cc = (int)(i % C);
  long long r = i / C;            // b*n + pointIdx
  if (cc < 3)       out[i] = p[r * 3 + cc];
  else if (cc == 3) out[i] = xin[r];
  else              out[i] = pf[r * 64 + (cc - 4)];
}

// h = concat(x4(512), lang(512), time(256)) broadcast over n -> (B,n,1280)
__global__ __launch_bounds__(256) void ctx_concat_kernel(
    const float* __restrict__ x4, const float* __restrict__ lang,
    const float* __restrict__ tm, float* __restrict__ h, int B, int n)
{
  const int C = 1280;
  long long i = (long long)blockIdx.x * 256 + threadIdx.x;
  if (i >= (long long)B * n * C) return;
  int cc = (int)(i % C);
  long long r = i / C;
  int b = (int)(r / n);
  if (cc < 512)       h[i] = x4[r * 512 + cc];
  else if (cc < 1024) h[i] = lang[(long long)b * 512 + (cc - 512)];
  else                h[i] = tm[(long long)b * 256 + (cc - 1024)];
}

// h = concat(x(B,n,c1), g(B,c2) broadcast)
__global__ __launch_bounds__(256) void cat_broadcast_kernel(
    const float* __restrict__ x, const float* __restrict__ g,
    float* __restrict__ h, int B, int n, int c1, int c2)
{
  int C = c1 + c2;
  long long i = (long long)blockIdx.x * 256 + threadIdx.x;
  if (i >= (long long)B * n * C) return;
  int cc = (int)(i % C);
  long long r = i / C;
  int b = (int)(r / n);
  h[i] = (cc < c1) ? x[r * c1 + cc] : g[(long long)b * c2 + (cc - c1)];
}

// =====================================================================
//                           Host orchestration
// =====================================================================
struct Lin { const float* w; const float* b; };
struct BNp { const float* g; const float* b; };
struct Block {
  BNp bn1, bn2, bn3; Lin kk; const float* lin1; const float* lin3;
  Lin p1, p2; BNp pbn; Lin q, v; Lin w1, w2; BNp wbn1, wbn2;
};
struct TDn { BNp bn; const float* w; };
struct TUp { BNp bn1, bn2; Lin l1, l2; };

struct Alloc {
  char* base; size_t off;
  float* f(size_t n) { float* r = (float*)(base + off); off += (n * sizeof(float) + 255) & ~(size_t)255; return r; }
  int*   i(size_t n) { int*   r = (int*)(base + off);   off += (n * sizeof(int)   + 255) & ~(size_t)255; return r; }
};

static inline unsigned nblk(long long tot) { return (unsigned)((tot + 255) / 256); }
static inline int cdiv(int a, int b) { return (a + b - 1) / b; }

static void run_gemm(hipStream_t st, const float* A, const float* W, const float* bias,
                     float* C, long long M, int K, int N, int relu) {
  dim3 g(cdiv(N, BN), (unsigned)((M + BM - 1) / BM));
  gemm_wmma_kernel<<<g, 256, 0, st>>>(A, W, bias, C, (int)M, K, N, relu);
}

static void run_bn(Alloc s, hipStream_t st, float* x, BNp bn, long long M, int C, int relu) {
  float* mv = s.f((size_t)2 * C);
  bn_stats_kernel<<<C, 256, 0, st>>>(x, (int)M, C, mv);
  bn_apply_kernel<<<nblk(M * C), 256, 0, st>>>(x, mv, bn.g, bn.b, (int)M, C, relu);
}

static void run_pt_block(Alloc s, hipStream_t st, const Block& bp, const float* p,
                         const float* x, float* out, int B, int n, int c, int ns) {
  long long M = (long long)B * n;
  long long Mn = M * ns;
  int cs = c / 8;
  float* y = s.f(M * c);
  run_gemm(st, x, bp.lin1, nullptr, y, M, c, c, 0);
  run_bn(s, st, y, bp.bn1, M, c, 1);
  float* qb = s.f(M * c); run_gemm(st, y, bp.q.w, bp.q.b, qb, M, c, c, 0);
  float* kb = s.f(M * c); run_gemm(st, y, bp.kk.w, bp.kk.b, kb, M, c, c, 0);
  float* vb = s.f(M * c); run_gemm(st, y, bp.v.w, bp.v.b, vb, M, c, c, 0);
  int* idx = s.i(Mn);
  if (ns == 8) knn_kernel<8><<<nblk(M), 256, 0, st>>>(p, p, idx, B, n, n);
  else         knn_kernel<16><<<nblk(M), 256, 0, st>>>(p, p, idx, B, n, n);
  float* kg = s.f(Mn * c);
  gather_kernel<<<nblk(Mn * c), 256, 0, st>>>(kb, idx, kg, B, n, ns, n, c);
  float* vg = s.f(Mn * c);
  gather_kernel<<<nblk(Mn * c), 256, 0, st>>>(vb, idx, vg, B, n, ns, n, c);
  float* pr3 = s.f(Mn * 3);   // relative coords (group_cat with c==0)
  group_cat_kernel<<<nblk(Mn * 3), 256, 0, st>>>(p, p, p, idx, pr3, B, n, ns, n, 0);
  float* t3 = s.f(Mn * 3);
  run_gemm(st, pr3, bp.p1.w, bp.p1.b, t3, Mn, 3, 3, 0);
  run_bn(s, st, t3, bp.pbn, Mn, 3, 1);
  float* pr = s.f(Mn * c);
  run_gemm(st, t3, bp.p2.w, bp.p2.b, pr, Mn, 3, c, 0);
  qkp_combine_kernel<<<nblk(Mn * c), 256, 0, st>>>(kg, qb, pr, (int)M, ns, c);
  run_bn(s, st, kg, bp.wbn1, Mn, c, 1);
  float* wA = s.f(Mn * cs);
  run_gemm(st, kg, bp.w1.w, bp.w1.b, wA, Mn, c, cs, 0);
  run_bn(s, st, wA, bp.wbn2, Mn, cs, 1);
  float* wS = s.f(Mn * cs);
  run_gemm(st, wA, bp.w2.w, bp.w2.b, wS, Mn, cs, cs, 0);
  softmax_ns_kernel<<<nblk(M * cs), 256, 0, st>>>(wS, (int)M, ns, cs);
  float* att = s.f(M * c);
  attn_out_kernel<<<nblk(M * c), 256, 0, st>>>(vg, pr, wS, att, (int)M, ns, c, cs);
  run_bn(s, st, att, bp.bn2, M, c, 1);
  run_gemm(st, att, bp.lin3, nullptr, out, M, c, c, 0);
  run_bn(s, st, out, bp.bn3, M, c, 0);
  add_kernel<<<nblk(M * c), 256, 0, st>>>(out, x, M * c, 1);
}

static void run_t_down(Alloc s, hipStream_t st, const TDn& td, const float* p, const float* x,
                       float* pOut, float* xOut, int B, int n, int cin, int stride, int ns, int cout) {
  int m = n / stride;
  long long M = (long long)B * m, Mn = M * ns;
  int* sidx = s.i(M);
  fps_kernel<<<B, 256, 0, st>>>(p, sidx, B, n, m);
  gather_kernel<<<nblk(M * 3), 256, 0, st>>>(p, sidx, pOut, B, m, 1, n, 3);
  int* nidx = s.i(Mn);
  knn_kernel<16><<<nblk(M), 256, 0, st>>>(pOut, p, nidx, B, m, n);
  int C = 3 + cin;
  float* grp = s.f(Mn * C);
  group_cat_kernel<<<nblk(Mn * C), 256, 0, st>>>(p, x, pOut, nidx, grp, B, m, ns, n, cin);
  float* g2 = s.f(Mn * cout);
  run_gemm(st, grp, td.w, nullptr, g2, Mn, C, cout, 0);
  run_bn(s, st, g2, td.bn, Mn, cout, 1);
  maxpool_ns_kernel<<<nblk(M * cout), 256, 0, st>>>(g2, xOut, (int)M, ns, cout);
}

static void run_t_up(Alloc s, hipStream_t st, const TUp& tu, const float* pf, const float* xf,
                     const float* pc, const float* xc, float* out,
                     int B, int nf, int cf, int nc, int cc, int co) {
  long long Mf = (long long)B * nf, Mc = (long long)B * nc;
  float* a = s.f(Mf * co);
  run_gemm(st, xf, tu.l1.w, tu.l1.b, a, Mf, cf, co, 0);
  run_bn(s, st, a, tu.bn1, Mf, co, 1);
  float* bb = s.f(Mc * co);
  run_gemm(st, xc, tu.l2.w, tu.l2.b, bb, Mc, cc, co, 0);
  run_bn(s, st, bb, tu.bn2, Mc, co, 1);
  interp_kernel<<<nblk(Mf), 256, 0, st>>>(pf, pc, bb, out, B, nf, nc, co);
  add_kernel<<<nblk(Mf * co), 256, 0, st>>>(out, a, Mf * co, 0);
}

extern "C" void kernel_launch(void* const* d_in, const int* in_sizes, int n_in,
                              void* d_out, int out_size, void* d_ws, size_t ws_size,
                              hipStream_t stream) {
  const int B = 4, N = 4096;
  hipStream_t st = stream;

  // -------- input layout resolution --------
  // Layout A (top-level insertion order): params leaves [0..271], then
  //   x, point_feat, language_feat, time_embedding, c_pc_xyz.
  // Layout B (fully sorted): c_pc_xyz, language_feat, params [2..273],
  //   point_feat, time_embedding, x.
  const float *in_x, *in_pf, *in_lang, *in_time, *in_p;
  int pi;
  if (in_sizes[0] == B * N * 3) {           // layout B
    in_p = (const float*)d_in[0];
    in_lang = (const float*)d_in[1];
    pi = 2;
    in_pf = (const float*)d_in[n_in - 3];
    in_time = (const float*)d_in[n_in - 2];
    in_x = (const float*)d_in[n_in - 1];
  } else {                                   // layout A
    pi = 0;
    in_x = (const float*)d_in[n_in - 5];
    in_pf = (const float*)d_in[n_in - 4];
    in_lang = (const float*)d_in[n_in - 3];
    in_time = (const float*)d_in[n_in - 2];
    in_p = (const float*)d_in[n_in - 1];
  }

  auto nf = [&]() { return (const float*)d_in[pi++]; };
  auto rdLin = [&](Lin& l) { l.b = nf(); l.w = nf(); };     // sorted: 'b' < 'w'
  auto rdBN  = [&](BNp& b) { b.b = nf(); b.g = nf(); };     // sorted: 'b' < 'g'
  auto rdBlock = [&](Block& bl) {                           // sorted block_p keys
    rdBN(bl.bn1); rdBN(bl.bn2); rdBN(bl.bn3);
    rdLin(bl.kk); bl.lin1 = nf(); bl.lin3 = nf();
    rdLin(bl.p1); rdLin(bl.p2); rdBN(bl.pbn);
    rdLin(bl.q);  rdLin(bl.v);
    rdLin(bl.w1); rdLin(bl.w2);
    rdBN(bl.wbn1); rdBN(bl.wbn2);
  };
  auto rdTU = [&](TUp& t, bool bn2) { rdBN(t.bn1); if (bn2) rdBN(t.bn2); rdLin(t.l1); rdLin(t.l2); };
  auto rdTD = [&](TDn& t) { rdBN(t.bn); t.w = nf(); };

  Lin ctx1, ctx2; BNp ctx_bn;
  rdLin(ctx1); rdLin(ctx2); rdBN(ctx_bn);
  Block dec1_b0; rdBlock(dec1_b0); TUp dec1_tu; rdTU(dec1_tu, true);
  Block dec2_b0; rdBlock(dec2_b0); TUp dec2_tu; rdTU(dec2_tu, true);
  Block dec3_b0; rdBlock(dec3_b0); TUp dec3_tu; rdTU(dec3_tu, true);
  Block dec4_b0; rdBlock(dec4_b0); TUp dec4_tu; rdTU(dec4_tu, false);
  Block enc1_b0; rdBlock(enc1_b0); TDn enc1_td; rdTD(enc1_td);
  Block enc2_b0; rdBlock(enc2_b0); TDn enc2_td; rdTD(enc2_td);
  Block enc3_b0; rdBlock(enc3_b0); TDn enc3_td; rdTD(enc3_td);
  Block enc4_b0; rdBlock(enc4_b0); TDn enc4_td; rdTD(enc4_td);

  Alloc al{(char*)d_ws, 0};

  // -------------------- encoder --------------------
  float* x0 = al.f((size_t)B * N * 68);
  build_x0_kernel<<<nblk((long long)B * N * 68), 256, 0, st>>>(in_p, in_x, in_pf, x0, B, N);

  const float* p1 = in_p;
  float* x1 = al.f((size_t)B * N * 64);
  { Alloc s = al;
    run_gemm(st, x0, enc1_td.w, nullptr, x1, (long long)B * N, 68, 64, 0);
    run_bn(s, st, x1, enc1_td.bn, (long long)B * N, 64, 1); }
  float* x1e = al.f((size_t)B * N * 64);
  { Alloc s = al; run_pt_block(s, st, enc1_b0, p1, x1, x1e, B, N, 64, 8); }

  float* p2 = al.f((size_t)B * 1024 * 3);
  float* x2 = al.f((size_t)B * 1024 * 128);
  { Alloc s = al; run_t_down(s, st, enc2_td, p1, x1e, p2, x2, B, 4096, 64, 4, 16, 128); }
  float* x2e = al.f((size_t)B * 1024 * 128);
  { Alloc s = al; run_pt_block(s, st, enc2_b0, p2, x2, x2e, B, 1024, 128, 16); }

  float* p3 = al.f((size_t)B * 256 * 3);
  float* x3 = al.f((size_t)B * 256 * 256);
  { Alloc s = al; run_t_down(s, st, enc3_td, p2, x2e, p3, x3, B, 1024, 128, 4, 16, 256); }
  float* x3e = al.f((size_t)B * 256 * 256);
  { Alloc s = al; run_pt_block(s, st, enc3_b0, p3, x3, x3e, B, 256, 256, 16); }

  float* p4 = al.f((size_t)B * 64 * 3);
  float* x4 = al.f((size_t)B * 64 * 512);
  { Alloc s = al; run_t_down(s, st, enc4_td, p3, x3e, p4, x4, B, 256, 256, 4, 16, 512); }
  float* x4e = al.f((size_t)B * 64 * 512);
  { Alloc s = al; run_pt_block(s, st, enc4_b0, p4, x4, x4e, B, 64, 512, 16); }

  // -------------------- context fusion --------------------
  float* x4c = al.f((size_t)B * 64 * 512);
  { Alloc s = al;
    float* h = s.f((size_t)B * 64 * 1280);
    ctx_concat_kernel<<<nblk((long long)B * 64 * 1280), 256, 0, st>>>(x4e, in_lang, in_time, h, B, 64);
    float* t = s.f((size_t)B * 64 * 512);
    run_gemm(st, h, ctx1.w, ctx1.b, t, (long long)B * 64, 1280, 512, 0);
    run_bn(s, st, t, ctx_bn, (long long)B * 64, 512, 1);
    run_gemm(st, t, ctx2.w, ctx2.b, x4c, (long long)B * 64, 512, 512, 0); }

  // -------------------- dec4 head --------------------
  float* x4d = al.f((size_t)B * 64 * 512);
  { Alloc s = al;
    float* xm = s.f((size_t)B * 512);
    mean_axis1_kernel<<<nblk(B * 512), 256, 0, st>>>(x4c, xm, B, 64, 512);
    float* g = s.f((size_t)B * 512);
    run_gemm(st, xm, dec4_tu.l2.w, dec4_tu.l2.b, g, B, 512, 512, 1);
    float* h = s.f((size_t)B * 64 * 1024);
    cat_broadcast_kernel<<<nblk((long long)B * 64 * 1024), 256, 0, st>>>(x4c, g, h, B, 64, 512, 512);
    run_gemm(st, h, dec4_tu.l1.w, dec4_tu.l1.b, x4d, (long long)B * 64, 1024, 512, 0);
    run_bn(s, st, x4d, dec4_tu.bn1, (long long)B * 64, 512, 1); }
  float* x4dd = al.f((size_t)B * 64 * 512);
  { Alloc s = al; run_pt_block(s, st, dec4_b0, p4, x4d, x4dd, B, 64, 512, 16); }

  // -------------------- decoder --------------------
  float* x3u = al.f((size_t)B * 256 * 256);
  { Alloc s = al; run_t_up(s, st, dec3_tu, p3, x3e, p4, x4dd, x3u, B, 256, 256, 64, 512, 256); }
  float* x3d = al.f((size_t)B * 256 * 256);
  { Alloc s = al; run_pt_block(s, st, dec3_b0, p3, x3u, x3d, B, 256, 256, 16); }

  float* x2u = al.f((size_t)B * 1024 * 128);
  { Alloc s = al; run_t_up(s, st, dec2_tu, p2, x2e, p3, x3d, x2u, B, 1024, 128, 256, 256, 128); }
  float* x2d = al.f((size_t)B * 1024 * 128);
  { Alloc s = al; run_pt_block(s, st, dec2_b0, p2, x2u, x2d, B, 1024, 128, 16); }

  float* x1u = al.f((size_t)B * 4096 * 64);
  { Alloc s = al; run_t_up(s, st, dec1_tu, p1, x1e, p2, x2d, x1u, B, 4096, 64, 1024, 128, 64); }
  { Alloc s = al; run_pt_block(s, st, dec1_b0, p1, x1u, (float*)d_out, B, 4096, 64, 8); }

  (void)out_size; (void)ws_size;
}
